// Decoder_TRANSFORMERCLUSTER_6751688589496
// MI455X (gfx1250) — matrix operand order repaired and checked by reading, not compiled, not run on hardware
//
#include <hip/hip_runtime.h>

// ---------------- constants from the reference ----------------
#define B_        64
#define D_        512
#define H_        8
#define L_        1024
#define NJ_       25
#define NF_       6
#define MAX_ACT_  6
#define CLUSTERS_ 10
#define KM_ITERS_ 10
#define NLAYERS_  6
#define FF_       2048
#define EH_       64            // D_/H_
#define M_        (B_ * L_)     // 65536 rows for the big GEMMs

// ---------------- vector types for WMMA ----------------
typedef __attribute__((ext_vector_type(8)))  float          v8f;
typedef __attribute__((ext_vector_type(16))) __bf16         bf16x16;
typedef __attribute__((ext_vector_type(8)))  unsigned short u16x8;

union FragU { u16x8 h[2]; bf16x16 v; };

__device__ __forceinline__ unsigned short f32_to_bf16(float f) {
  unsigned int u = __float_as_uint(f);
  unsigned int r = u + 0x7FFFu + ((u >> 16) & 1u); // round-to-nearest-even
  return (unsigned short)(r >> 16);
}

// CDNA5 async global->LDS copy (ASYNCcnt-tracked, bypasses VGPRs).
// lds = low 32 bits of generic LDS pointer (hardware truncates flat addr).
__device__ __forceinline__ void async_copy_b128(unsigned lds, const void* gaddr) {
  asm volatile("global_load_async_to_lds_b128 %0, %1, off"
               :: "v"(lds), "v"(gaddr) : "memory");
}

// ---------------- elementwise f32 -> bf16 ----------------
__global__ void conv_bf16_kernel(const float* __restrict__ in,
                                 unsigned short* __restrict__ out, size_t n) {
  size_t i = (size_t)blockIdx.x * blockDim.x + threadIdx.x;
  if (i < n) out[i] = f32_to_bf16(in[i]);
}

// ---------------- W(K,N) f32 -> Wt(N,K) bf16 ----------------
__global__ void transpose_bf16_kernel(const float* __restrict__ W,
                                      unsigned short* __restrict__ Wt,
                                      int K, int N) {
  int idx = blockIdx.x * blockDim.x + threadIdx.x;
  if (idx >= K * N) return;
  int n = idx / K;
  int k = idx - n * K;
  Wt[idx] = f32_to_bf16(W[(size_t)k * N + n]);
}

// ---------------- positional encoding: x = pe(L,D) broadcast over B ----------
__global__ void pe_init_kernel(float* __restrict__ x, unsigned short* __restrict__ xb) {
  int i = blockIdx.x * blockDim.x + threadIdx.x;  // over L_*D_
  if (i >= L_ * D_) return;
  int l = i / D_, d = i - l * D_;
  int pair = d >> 1;
  float div = __expf((float)(2 * pair) * (-9.210340371976184f / (float)D_)); // -ln(1e4)/D
  float ang = (float)l * div;
  float val = (d & 1) ? __cosf(ang) : __sinf(ang);
  unsigned short bv = f32_to_bf16(val);
  for (int b = 0; b < B_; ++b) {
    size_t off = ((size_t)b * L_ + l) * D_ + d;
    x[off] = val;
    xb[off] = bv;
  }
}

// ---------------- abflat[b, a*D+d] = actionBiases[y[b,a], d] ----------------
__global__ void gather_ab_kernel(const int* __restrict__ y,
                                 const float* __restrict__ ab,
                                 float* __restrict__ abflat,
                                 unsigned short* __restrict__ abflat_b) {
  int i = blockIdx.x * blockDim.x + threadIdx.x;  // B_*MAX_ACT_*D_
  if (i >= B_ * MAX_ACT_ * D_) return;
  int d = i % D_;
  int a = (i / D_) % MAX_ACT_;
  int b = i / (D_ * MAX_ACT_);
  float v = ab[(size_t)y[b * MAX_ACT_ + a] * D_ + d];
  abflat[i] = v;
  abflat_b[i] = f32_to_bf16(v);
}

// ---------------- WMMA bf16 GEMM: C = A(MxK) @ Bt(NxK)^T + bias (+res)(+relu) ---
// Tiles staged with CDNA5 async global->LDS copies, double-buffered.
// Edge handling: row-clamped loads (OOB A rows / Bt rows feed only never-stored
// C rows/cols), stores guarded.
__global__ __launch_bounds__(256)
void wmma_gemm_kernel(const unsigned short* __restrict__ A,
                      const unsigned short* __restrict__ Bt,
                      const float* __restrict__ bias,
                      const float* __restrict__ res,
                      float* __restrict__ C,
                      int M, int N, int K, int relu) {
  constexpr int TM = 128, TN = 128, TK = 32, LDT = TK + 8;
  __shared__ unsigned short sA[2][TM * LDT];   // 2 x 10 KB
  __shared__ unsigned short sB[2][TN * LDT];   // 2 x 10 KB

  const int tid  = threadIdx.x;
  const int lane = tid & 31;
  const int wave = tid >> 5;
  const int wm = (wave >> 1) * 32;   // 4 waves along M
  const int wn = (wave & 1) * 64;    // 2 waves along N
  const int m0 = blockIdx.y * TM;
  const int n0 = blockIdx.x * TN;
  const int hl = lane & 15;
  const int hi = lane >> 4;

  v8f acc[2][4] = {};

  // staging: thread -> (row, 16-half chunk)
  const int ldRow = tid >> 1;          // 0..127
  const int ldCol = (tid & 1) << 4;    // 0 or 16
  int gm = m0 + ldRow; if (gm >= M) gm = M - 1;   // clamp (exact, see note)
  int gn = n0 + ldRow; if (gn >= N) gn = N - 1;
  const unsigned short* gA = A  + (size_t)gm * K + ldCol;
  const unsigned short* gB = Bt + (size_t)gn * K + ldCol;

  const unsigned ldsA0 = (unsigned)(uintptr_t)&sA[0][ldRow * LDT + ldCol];
  const unsigned ldsB0 = (unsigned)(uintptr_t)&sB[0][ldRow * LDT + ldCol];
  const unsigned abufBytes = (unsigned)(TM * LDT * sizeof(unsigned short)); // 10240
  const unsigned bbufBytes = (unsigned)(TN * LDT * sizeof(unsigned short));

  // prologue: tile 0 -> buffer 0 (4 async b128 per thread)
  async_copy_b128(ldsA0,      gA);
  async_copy_b128(ldsA0 + 16, gA + 8);
  async_copy_b128(ldsB0,      gB);
  async_copy_b128(ldsB0 + 16, gB + 8);

  int buf = 0;
  for (int k0 = 0; k0 < K; k0 += TK) {
    if (k0 + TK < K) {
      // issue tile k+1 into the other buffer, then wait for tile k (oldest 4)
      const unsigned short* pA = gA + (k0 + TK);
      const unsigned short* pB = gB + (k0 + TK);
      unsigned la = ldsA0 + (unsigned)(buf ^ 1) * abufBytes;
      unsigned lb = ldsB0 + (unsigned)(buf ^ 1) * bbufBytes;
      async_copy_b128(la,      pA);
      async_copy_b128(la + 16, pA + 8);
      async_copy_b128(lb,      pB);
      async_copy_b128(lb + 16, pB + 8);
      asm volatile("s_wait_asynccnt 0x4" ::: "memory");
    } else {
      asm volatile("s_wait_asynccnt 0x0" ::: "memory");
    }
    __syncthreads();

    const unsigned short* bufA = sA[buf];
    const unsigned short* bufB = sB[buf];

    FragU afr[2], bfr[4];
#pragma unroll
    for (int mt = 0; mt < 2; ++mt) {
      int r = wm + mt * 16 + hl;
      // 16-bit A 16x32 layout: lanes 0-15 K=0..7,16..23 ; lanes 16-31 K=8..15,24..31
      afr[mt].h[0] = *(const u16x8*)&bufA[r * LDT + hi * 8];
      afr[mt].h[1] = *(const u16x8*)&bufA[r * LDT + 16 + hi * 8];
    }
#pragma unroll
    for (int nt = 0; nt < 4; ++nt) {
      int c = wn + nt * 16 + hl;
      // B 32x16 layout: lanes 0-15 K=0..7,16..23 ; lanes 16-31 K=8..15,24..31 (matches A)
      bfr[nt].h[0] = *(const u16x8*)&bufB[c * LDT + hi * 8];
      bfr[nt].h[1] = *(const u16x8*)&bufB[c * LDT + 16 + hi * 8];
    }
#pragma unroll
    for (int mt = 0; mt < 2; ++mt)
#pragma unroll
      for (int nt = 0; nt < 4; ++nt)
        acc[mt][nt] = __builtin_amdgcn_wmma_f32_16x16x32_bf16(
            false, afr[mt].v, false, bfr[nt].v, (short)0, acc[mt][nt], false, false);
    __syncthreads();   // all waves done reading `buf` before it is re-filled
    buf ^= 1;
  }

  // C/D layout: VGPR i -> row = hi*8 + i, col = lane&15
#pragma unroll
  for (int mt = 0; mt < 2; ++mt) {
#pragma unroll
    for (int nt = 0; nt < 4; ++nt) {
      int col = n0 + wn + nt * 16 + hl;
      if (col >= N) continue;
      float bval = bias ? bias[col] : 0.f;
#pragma unroll
      for (int i = 0; i < 8; ++i) {
        int row = m0 + wm + mt * 16 + hi * 8 + i;
        if (row >= M) continue;
        float val = acc[mt][nt][i] + bval;
        if (res) val += res[(size_t)row * N + col];
        if (relu) val = fmaxf(val, 0.f);
        C[(size_t)row * N + col] = val;
      }
    }
  }
}

// ---------------- fused k-means clustered self-attention -------------------
// one block per (b,h) group; q,k,v,o are (B,L,D) with head offset h*EH_
__global__ __launch_bounds__(256)
void kmeans_attn_kernel(const float* __restrict__ q, const float* __restrict__ k,
                        const float* __restrict__ v, float* __restrict__ o) {
  const int g = blockIdx.x;
  const int b = g / H_, h = g - b * H_;
  const float* qg = q + ((size_t)b * L_) * D_ + h * EH_;
  const float* kg = k + ((size_t)b * L_) * D_ + h * EH_;
  const float* vg = v + ((size_t)b * L_) * D_ + h * EH_;
  float*       og = o + ((size_t)b * L_) * D_ + h * EH_;
  const int tid = threadIdx.x;

  __shared__ float cent[CLUSTERS_][EH_];
  __shared__ float csum[CLUSTERS_][EH_];
  __shared__ float ccnt[CLUSTERS_];
  __shared__ float cq2[CLUSTERS_];
  __shared__ float sc[CLUSTERS_][L_];   // 40 KB scores
  __shared__ float oc[CLUSTERS_][EH_];
  __shared__ float red[256];

  for (int i = tid; i < CLUSTERS_ * EH_; i += 256)
    cent[i / EH_][i % EH_] = qg[(size_t)(i / EH_) * D_ + (i % EH_)];
  __syncthreads();

  int myAssign[L_ / 256];

  for (int it = 0; it < KM_ITERS_; ++it) {
    for (int i = tid; i < CLUSTERS_ * EH_; i += 256) csum[i / EH_][i % EH_] = 0.f;
    if (tid < CLUSTERS_) {
      ccnt[tid] = 0.f;
      float s = 0.f;
#pragma unroll
      for (int e = 0; e < EH_; ++e) s += cent[tid][e] * cent[tid][e];
      cq2[tid] = s;
    }
    __syncthreads();

#pragma unroll
    for (int r = 0; r < L_ / 256; ++r) {
      int l = r * 256 + tid;
      const float* qr = qg + (size_t)l * D_;
      float qv[EH_];
#pragma unroll
      for (int e = 0; e < EH_; ++e) qv[e] = qr[e];
      float best = 3.4e38f; int bi = 0;
#pragma unroll
      for (int c = 0; c < CLUSTERS_; ++c) {
        float dot = 0.f;
#pragma unroll
        for (int e = 0; e < EH_; ++e) dot += qv[e] * cent[c][e];
        float dist = cq2[c] - 2.f * dot;   // ||q||^2 constant -> drop
        if (dist < best) { best = dist; bi = c; }
      }
      myAssign[r] = bi;
      atomicAdd(&ccnt[bi], 1.f);
#pragma unroll
      for (int e = 0; e < EH_; ++e) atomicAdd(&csum[bi][e], qv[e]);
    }
    __syncthreads();

    for (int i = tid; i < CLUSTERS_ * EH_; i += 256) {
      int c = i / EH_, e = i - c * EH_;
      float cnt = ccnt[c];
      if (cnt > 0.f) cent[c][e] = csum[c][e] / cnt;
    }
    __syncthreads();
  }

  // scores = cent @ k^T / sqrt(E)
  const float scale = 0.125f;
#pragma unroll
  for (int r = 0; r < L_ / 256; ++r) {
    int s = r * 256 + tid;
    const float* kr = kg + (size_t)s * D_;
    float kv[EH_];
#pragma unroll
    for (int e = 0; e < EH_; ++e) kv[e] = kr[e];
#pragma unroll
    for (int c = 0; c < CLUSTERS_; ++c) {
      float dot = 0.f;
#pragma unroll
      for (int e = 0; e < EH_; ++e) dot += cent[c][e] * kv[e];
      sc[c][s] = dot * scale;
    }
  }
  __syncthreads();

  // softmax over s for each cluster row
  for (int c = 0; c < CLUSTERS_; ++c) {
    float m = -3.4e38f;
    for (int s = tid; s < L_; s += 256) m = fmaxf(m, sc[c][s]);
    red[tid] = m; __syncthreads();
    for (int o2 = 128; o2 > 0; o2 >>= 1) {
      if (tid < o2) red[tid] = fmaxf(red[tid], red[tid + o2]);
      __syncthreads();
    }
    float mx = red[0]; __syncthreads();
    float sum = 0.f;
    for (int s = tid; s < L_; s += 256) {
      float e1 = __expf(sc[c][s] - mx);
      sc[c][s] = e1;
      sum += e1;
    }
    red[tid] = sum; __syncthreads();
    for (int o2 = 128; o2 > 0; o2 >>= 1) {
      if (tid < o2) red[tid] += red[tid + o2];
      __syncthreads();
    }
    float inv = 1.f / red[0]; __syncthreads();
    for (int s = tid; s < L_; s += 256) sc[c][s] *= inv;
    __syncthreads();
  }

  // oc[c][e] = sum_s A[c][s] * v[s][e]
  for (int i = tid; i < CLUSTERS_ * EH_; i += 256) {
    int c = i / EH_, e = i - c * EH_;
    float acc = 0.f;
    for (int s = 0; s < L_; ++s) acc += sc[c][s] * vg[(size_t)s * D_ + e];
    oc[c][e] = acc;
  }
  __syncthreads();

  // gather: o[l] = oc[assign[l]]
#pragma unroll
  for (int r = 0; r < L_ / 256; ++r) {
    int l = r * 256 + tid;
    int c = myAssign[r];
    float* orow = og + (size_t)l * D_;
#pragma unroll
    for (int e = 0; e < EH_; ++e) orow[e] = oc[c][e];
  }
}

// ---------------- LayerNorm (optionally adds per-batch broadcast row) -------
// out = g*(v-mean)/sqrt(var+eps)+b where v = src[row] (+ bcast[row/L] if given)
__global__ __launch_bounds__(256)
void layernorm_kernel(const float* __restrict__ src, const float* __restrict__ bcast,
                      const float* __restrict__ gg, const float* __restrict__ bb,
                      float* __restrict__ xout, unsigned short* __restrict__ xbout) {
  const int row = blockIdx.x;
  const int tid = threadIdx.x;
  const float* s = src + (size_t)row * D_;
  const float* bc = bcast ? bcast + (size_t)(row / L_) * D_ : nullptr;
  float v0 = s[tid]       + (bc ? bc[tid]       : 0.f);
  float v1 = s[tid + 256] + (bc ? bc[tid + 256] : 0.f);

  __shared__ float red[256];
  red[tid] = v0 + v1; __syncthreads();
  for (int o = 128; o > 0; o >>= 1) { if (tid < o) red[tid] += red[tid + o]; __syncthreads(); }
  float mean = red[0] * (1.f / (float)D_); __syncthreads();
  float d0 = v0 - mean, d1 = v1 - mean;
  red[tid] = d0 * d0 + d1 * d1; __syncthreads();
  for (int o = 128; o > 0; o >>= 1) { if (tid < o) red[tid] += red[tid + o]; __syncthreads(); }
  float inv = rsqrtf(red[0] * (1.f / (float)D_) + 1e-5f);

  float o0 = gg[tid] * d0 * inv + bb[tid];
  float o1 = gg[tid + 256] * d1 * inv + bb[tid + 256];
  size_t base = (size_t)row * D_;
  xout[base + tid]        = o0;
  xout[base + tid + 256]  = o1;
  xbout[base + tid]       = f32_to_bf16(o0);
  xbout[base + tid + 256] = f32_to_bf16(o1);
}

// ---------------- final: out[b][j][f][l] = val[b*L+l][j*NF+f] * mask --------
__global__ void final_out_kernel(const float* __restrict__ val,
                                 const unsigned char* __restrict__ mask,
                                 float* __restrict__ out) {
  size_t i = (size_t)blockIdx.x * blockDim.x + threadIdx.x;
  const size_t total = (size_t)B_ * NJ_ * NF_ * L_;
  if (i >= total) return;
  int l = (int)(i % L_);
  int f = (int)((i / L_) % NF_);
  int j = (int)((i / ((size_t)L_ * NF_)) % NJ_);
  int b = (int)(i / ((size_t)L_ * NF_ * NJ_));
  float m = mask[(size_t)b * L_ + l] ? 1.f : 0.f;
  out[i] = val[((size_t)b * L_ + l) * (NJ_ * NF_) + j * NF_ + f] * m;
}

// ---------------- host-side helpers ----------------
static void run_conv(const float* in, unsigned short* out, size_t n, hipStream_t s) {
  unsigned grid = (unsigned)((n + 255) / 256);
  conv_bf16_kernel<<<grid, 256, 0, s>>>(in, out, n);
}

static void run_gemm(const unsigned short* A, const float* W, unsigned short* wslot,
                     const float* bias, const float* res, float* C,
                     int M, int N, int K, int relu, hipStream_t s) {
  int tot = K * N;
  transpose_bf16_kernel<<<(tot + 255) / 256, 256, 0, s>>>(W, wslot, K, N);
  dim3 grid((N + 127) / 128, (M + 127) / 128);
  wmma_gemm_kernel<<<grid, 256, 0, s>>>(A, wslot, bias, res, C, M, N, K, relu);
}

extern "C" void kernel_launch(void* const* d_in, const int* in_sizes, int n_in,
                              void* d_out, int out_size, void* d_ws, size_t ws_size,
                              hipStream_t stream) {
  (void)in_sizes; (void)n_in; (void)out_size; (void)ws_size;

  // input order: z, y, mask, lengths, then params leaves in jax tree order
  // (dict keys sorted alphabetically, lists in order):
  //   ab_b, ab_w, actionBiases, f_b, f_w,
  //   layers[0..5]: b1, b2, ca{bk,bo,bq,bv,wk,wo,wq,wv},
  //                 n1b, n1g, n2b, n2g, n3b, n3g,
  //                 sa{bk,bo,bq,bv,wk,wo,wq,wv}, w1, w2
  const float*         z    = (const float*)d_in[0];
  const int*           y    = (const int*)d_in[1];
  const unsigned char* mask = (const unsigned char*)d_in[2];
  const float* ab_b = (const float*)d_in[4];
  const float* ab_w = (const float*)d_in[5];
  const float* aB   = (const float*)d_in[6];
  const float* f_b  = (const float*)d_in[7];
  const float* f_w  = (const float*)d_in[8];
  auto LP = [&](int layer, int slot) { return (const float*)d_in[9 + layer * 26 + slot]; };
  // slots: 0 b1, 1 b2, 2 ca_bk, 3 ca_bo, 4 ca_bq, 5 ca_bv, 6 ca_wk, 7 ca_wo,
  //        8 ca_wq, 9 ca_wv, 10 n1b, 11 n1g, 12 n2b, 13 n2g, 14 n3b, 15 n3g,
  //        16 sa_bk, 17 sa_bo, 18 sa_bq, 19 sa_bv, 20 sa_wk, 21 sa_wo,
  //        22 sa_wq, 23 sa_wv, 24 w1, 25 w2

  // ---- workspace carve-up ----
  char* ws = (char*)d_ws;
  size_t off = 0;
  auto carve = [&](size_t bytes) {
    off = (off + 255) & ~(size_t)255;
    void* p = ws + off;
    off += bytes;
    return p;
  };
  const size_t MD  = (size_t)M_ * D_;    // 33.5M elems
  const size_t MF  = (size_t)M_ * FF_;

  float*          x     = (float*)carve(MD * 4);
  unsigned short* xb    = (unsigned short*)carve(MD * 2);
  float*          qbuf  = (float*)carve(MD * 4);   // |
  float*          kbuf  = (float*)carve(MD * 4);   // | contiguous: FFN hidden
  float*          vbuf  = (float*)carve(MD * 4);   // | (M x FF f32) aliases q..o
  float*          obuf  = (float*)carve(MD * 4);   // |
  float*          hf    = qbuf;                    // alias: q/k/v/o dead during FFN
  float*          val   = qbuf;                    // alias: final logits (M x 150)
  unsigned short* ob16  = (unsigned short*)carve(MD * 2);
  float*          tbuf  = (float*)carve(MD * 4);
  unsigned short* hb    = (unsigned short*)carve(MF * 2);
  float*          abflat   = (float*)carve((size_t)B_ * MAX_ACT_ * D_ * 4);
  unsigned short* abflat_b = (unsigned short*)carve((size_t)B_ * MAX_ACT_ * D_ * 2);
  float*          zz    = (float*)carve((size_t)B_ * D_ * 4);
  unsigned short* zzb   = (unsigned short*)carve((size_t)B_ * D_ * 2);
  float*          cav   = (float*)carve((size_t)B_ * D_ * 4);
  unsigned short* cavb  = (unsigned short*)carve((size_t)B_ * D_ * 2);
  float*          co    = (float*)carve((size_t)B_ * D_ * 4);
  unsigned short* wslot = (unsigned short*)carve((size_t)(MAX_ACT_ * D_) * D_ * 2); // 3MB

  // ---- x = positional encoding (f32 + bf16) ----
  pe_init_kernel<<<(L_ * D_ + 255) / 256, 256, 0, stream>>>(x, xb);

  // ---- zz = z + (actionBiases[y].reshape) @ ab_w + ab_b ----
  gather_ab_kernel<<<(B_ * MAX_ACT_ * D_ + 255) / 256, 256, 0, stream>>>(y, aB, abflat, abflat_b);
  run_gemm(abflat_b, ab_w, wslot, ab_b, z, zz, B_, D_, MAX_ACT_ * D_, 0, stream);
  run_conv(zz, zzb, (size_t)B_ * D_, stream);

  for (int l = 0; l < NLAYERS_; ++l) {
    // ---- self-attention ----
    run_gemm(xb, LP(l, 22), wslot, LP(l, 18), nullptr, qbuf, M_, D_, D_, 0, stream); // q
    run_gemm(xb, LP(l, 20), wslot, LP(l, 16), nullptr, kbuf, M_, D_, D_, 0, stream); // k
    run_gemm(xb, LP(l, 23), wslot, LP(l, 19), nullptr, vbuf, M_, D_, D_, 0, stream); // v
    kmeans_attn_kernel<<<B_ * H_, 256, 0, stream>>>(qbuf, kbuf, vbuf, obuf);
    run_conv(obuf, ob16, MD, stream);
    run_gemm(ob16, LP(l, 21), wslot, LP(l, 17), x, tbuf, M_, D_, D_, 0, stream);     // o@wo + x
    layernorm_kernel<<<M_, 256, 0, stream>>>(tbuf, nullptr, LP(l, 11), LP(l, 10), x, xb);

    // ---- cross-attention: S==1 => softmax==1 => output is (zz@wv+bv)@wo+bo
    //      broadcast over L (exact algebraic simplification of the reference)
    run_gemm(zzb, LP(l, 9), wslot, LP(l, 5), nullptr, cav, B_, D_, D_, 0, stream);
    run_conv(cav, cavb, (size_t)B_ * D_, stream);
    run_gemm(cavb, LP(l, 7), wslot, LP(l, 3), nullptr, co, B_, D_, D_, 0, stream);
    layernorm_kernel<<<M_, 256, 0, stream>>>(x, co, LP(l, 13), LP(l, 12), x, xb);

    // ---- FFN ----
    run_gemm(xb, LP(l, 24), wslot, LP(l, 0), nullptr, hf, M_, FF_, D_, 1, stream);   // relu
    run_conv(hf, hb, MF, stream);
    run_gemm(hb, LP(l, 25), wslot, LP(l, 1), x, tbuf, M_, D_, FF_, 0, stream);
    layernorm_kernel<<<M_, 256, 0, stream>>>(tbuf, nullptr, LP(l, 15), LP(l, 14), x, xb);
  }

  // ---- final projection (N=150 with tile guards) + transpose/mask ----
  run_gemm(xb, f_w, wslot, f_b, nullptr, val, M_, NJ_ * NF_, D_, 0, stream);
  const size_t total = (size_t)B_ * NJ_ * NF_ * L_;
  final_out_kernel<<<(unsigned)((total + 255) / 256), 256, 0, stream>>>(
      val, mask, (float*)d_out);
}
